// Attention_7499012899591
// MI455X (gfx1250) — matrix operand order repaired and checked by reading, not compiled, not run on hardware
//
#include <hip/hip_runtime.h>
#include <hip/hip_bf16.h>
#include <cstdint>

#define B_  4
#define T_  2048
#define C_  768
#define H_  12
#define HD_ 64

typedef __attribute__((ext_vector_type(16))) __bf16 v16bf;
typedef __attribute__((ext_vector_type(8)))  __bf16 v8bf;
typedef __attribute__((ext_vector_type(8)))  float  v8f;

#if defined(__HIP_DEVICE_COMPILE__) && __has_builtin(__builtin_amdgcn_tensor_load_to_lds)
#define HAVE_TDM 1
#else
#define HAVE_TDM 0
#endif

// D = A(16x32 bf16) * B(32x16 bf16) + C(16x16 f32)
static __device__ __forceinline__ v8f wmma_bf16(v16bf a, v16bf b, v8f c) {
  return __builtin_amdgcn_wmma_f32_16x16x32_bf16(
      /*neg_a=*/false, a, /*neg_b=*/false, b,
      /*c_mod=*/(short)0, c, /*reuse_a=*/false, /*reuse_b=*/false);
}

// Build a 16-element bf16 fragment from two contiguous 8-element (16B) runs.
static __device__ __forceinline__ v16bf frag16(const __bf16* p0, const __bf16* p1) {
  v8bf lo = *(const v8bf*)p0;
  v8bf hi = *(const v8bf*)p1;
  return __builtin_shufflevector(lo, hi, 0,1,2,3,4,5,6,7,8,9,10,11,12,13,14,15);
}

// Async 16B global -> LDS copy (per-lane addresses). ASYNCcnt-tracked.
// LDS byte offset = low 32 bits of the generic pointer (SHARED aperture scheme).
static __device__ __forceinline__ void async_copy16(const __bf16* lds_dst, const __bf16* gsrc) {
  uint32_t lds = (uint32_t)(uintptr_t)lds_dst;
  uint64_t ga  = (uint64_t)(uintptr_t)gsrc;
  asm volatile("global_load_async_to_lds_b128 %0, %1, off" :: "v"(lds), "v"(ga) : "memory");
}
static __device__ __forceinline__ void wait_async_le1() {
  asm volatile("s_wait_asynccnt 0x1" ::: "memory");
}
static __device__ __forceinline__ void wait_async_0() {
  asm volatile("s_wait_asynccnt 0x0" ::: "memory");
}

#if HAVE_TDM
typedef __attribute__((ext_vector_type(4))) unsigned int u32x4;
typedef __attribute__((ext_vector_type(8))) int          i32x8;
typedef __attribute__((ext_vector_type(4))) int          i32x4;

// TDM: load a 32x64 bf16 tile (row stride 64 elems in memory) into LDS with a
// 16B pad after each 128B row => LDS row stride 144B (72 elems).
// D# layout per CDNA5 ISA ch.8 (group0: count/lds/global/type; group1: sizes).
static __device__ __forceinline__ void tdm_load_k_tile(
    const __bf16* lds_dst, const __bf16* gsrc, uint32_t rows_avail) {
  uint64_t ga  = (uint64_t)(uintptr_t)gsrc;
  uint32_t lds = (uint32_t)(uintptr_t)lds_dst;
  u32x4 g0 = {0, 0, 0, 0};
  g0.x = 1u;                                                 // count = 1 (valid D#)
  g0.y = lds;                                                // lds_addr (bytes)
  g0.z = (uint32_t)ga;                                       // global_addr[31:0]
  g0.w = (uint32_t)((ga >> 32) & 0x01ffffffu) | (2u << 30);  // addr[56:32] | type=2
  i32x8 g1 = {0, 0, 0, 0, 0, 0, 0, 0};
  // data_size=1 (2B) | pad_enable | pad_interval=4 (32 DW = 128B) | pad_amount=3 (4 DW = 16B)
  g1[0] = (int)((1u << 16) | (1u << 20) | (4u << 22) | (3u << 25));
  g1[1] = (int)((64u & 0xffffu) << 16);                      // tensor_dim0[15:0] = 64
  g1[2] = (int)(((64u >> 16) & 0xffffu) | ((rows_avail & 0xffffu) << 16)); // dim0 hi | dim1 lo
  g1[3] = (int)(((rows_avail >> 16) & 0xffffu) | (64u << 16));             // dim1 hi | tile_dim0=64
  g1[4] = (int)32;                                           // tile_dim1 = 32 rows
  g1[5] = (int)64;                                           // tensor_dim0_stride = 64 elems
  i32x4 z4 = {0, 0, 0, 0};
#if defined(__clang_major__) && (__clang_major__ >= 23)
  i32x8 z8 = {0, 0, 0, 0, 0, 0, 0, 0};
  __builtin_amdgcn_tensor_load_to_lds(g0, g1, z4, z4, z8, 0);
#else
  __builtin_amdgcn_tensor_load_to_lds(g0, g1, z4, z4, 0);
#endif
}
#endif

__global__ void cvt_bf16_kernel(__bf16* __restrict__ dst, const float* __restrict__ src, int n) {
  int i = blockIdx.x * blockDim.x + threadIdx.x;
  if (i < n) dst[i] = (__bf16)src[i];
}

// ---------------------------------------------------------------------------
// GEMM mainloop pattern (QKV + proj): block 256 threads (8 waves),
// block tile 256M x 64N, wave tile 32M x 64N (2 A-frags x 4 B-frags = 8 WMMA).
// W tile double-buffered in LDS via async copies (ASYNCcnt).
// ---------------------------------------------------------------------------
__global__ __launch_bounds__(256) void qkv_gemm_kernel(
    const __bf16* __restrict__ X, const __bf16* __restrict__ W,
    const float* __restrict__ bias,
    __bf16* __restrict__ Q, __bf16* __restrict__ Kb, __bf16* __restrict__ V) {
  __shared__ __bf16 sB[2][64 * 40];   // [n][k] tile, row stride 40 elems (80B)

  const int tid  = threadIdx.x;
  const int lane = tid & 31;
  const int w    = tid >> 5;
  const int half = lane >> 4;
  const int n15  = lane & 15;
  const int kb8  = half * 8;
  const int kb16 = half * 16;

  const int m0 = blockIdx.x * 256 + w * 32;
  const int n0 = blockIdx.y * 64;

  const int brow = tid >> 2;          // n-local 0..63
  const int bcol = (tid & 3) * 8;     // k-local 0,8,16,24
  const __bf16* wsrc = W + (size_t)(n0 + brow) * C_ + bcol;

  async_copy16(&sB[0][brow * 40 + bcol], wsrc);            // prologue: chunk 0 -> buf 0

  const __bf16* arow0 = X + (size_t)(m0 + n15) * C_;
  const __bf16* arow1 = X + (size_t)(m0 + 16 + n15) * C_;
  v8f acc[2][4] = {};

  for (int kc = 0; kc < C_; kc += 32) {
    const int cur = (kc >> 5) & 1;
    v16bf a0 = frag16(arow0 + kc + kb16, arow0 + kc + kb16 + 8);
    v16bf a1 = frag16(arow1 + kc + kb16, arow1 + kc + kb16 + 8);
    if (kc + 32 < C_) {
      async_copy16(&sB[cur ^ 1][brow * 40 + bcol], wsrc + kc + 32);
      wait_async_le1();   // in-order completion => current chunk has landed
    } else {
      wait_async_0();
    }
    __syncthreads();
    const __bf16* bb = &sB[cur][0];
    v16bf bf[4];
    #pragma unroll
    for (int nt = 0; nt < 4; ++nt) {
      const __bf16* bp = bb + (nt * 16 + n15) * 40 + kb16;
      bf[nt] = frag16(bp, bp + 8);
    }
    #pragma unroll
    for (int nt = 0; nt < 4; ++nt) {
      acc[0][nt] = wmma_bf16(a0, bf[nt], acc[0][nt]);
      acc[1][nt] = wmma_bf16(a1, bf[nt], acc[1][nt]);
    }
    __syncthreads();
  }

  const int which = blockIdx.y / 12;   // 0=q, 1=k, 2=v  (768/64 == 12)
  const int h     = blockIdx.y % 12;
  __bf16* dst = (which == 0) ? Q : (which == 1) ? Kb : V;
  #pragma unroll
  for (int ms = 0; ms < 2; ++ms) {
    #pragma unroll
    for (int nt = 0; nt < 4; ++nt) {
      float bv = bias[n0 + nt * 16 + n15];
      #pragma unroll
      for (int r = 0; r < 8; ++r) {
        int m = m0 + ms * 16 + r + half * 8;
        int b = m >> 11;
        int t = m & (T_ - 1);
        int d = nt * 16 + n15;
        dst[(((size_t)(b * H_ + h)) * T_ + t) * HD_ + d] = (__bf16)(acc[ms][nt][r] + bv);
      }
    }
  }
}

// ---------------------------------------------------------------------------
// Flash attention: per (b,h), causal, online softmax.
// Block: 256 threads (8 waves) = 128 query rows; wave owns 16 q rows.
// K tile: TDM tensor_load_to_lds (with LDS row padding) when available,
// else async copies. V staged transposed; P staged per-wave.
// ---------------------------------------------------------------------------
__global__ __launch_bounds__(256) void flash_attn_kernel(
    const __bf16* __restrict__ Q, const __bf16* __restrict__ K,
    const __bf16* __restrict__ V, __bf16* __restrict__ Y) {
  __shared__ __bf16 sK[32 * 72];        // [kv][d], row stride 72 (144B)
  __shared__ __bf16 sV[HD_ * 40];       // transposed: [d][kv], row stride 40
  __shared__ __bf16 sP[8 * 16 * 40];    // per-wave P: [q][kv], row stride 40

  const int tid  = threadIdx.x;
  const int lane = tid & 31;
  const int w    = tid >> 5;
  const int half = lane >> 4;
  const int n15  = lane & 15;
  const int kb8  = half * 8;
  const int kb16 = half * 16;

  const size_t base  = (size_t)blockIdx.y * T_ * HD_;   // (b*H + h) * T * hd
  const int blockQ0  = blockIdx.x * 128;
  const int waveQ0   = blockQ0 + w * 16;

  // Q A-fragments, resident for the whole kernel (hd=64 -> two 16x32 fragments)
  const __bf16* qrow = Q + base + (size_t)(waveQ0 + n15) * HD_;
  const v16bf aq0 = frag16(qrow +  0 + kb16, qrow +  0 + kb16 + 8);
  const v16bf aq1 = frag16(qrow + 32 + kb16, qrow + 32 + kb16 + 8);

  v8f yacc[4] = {};
  float mi[8], li[8];
  #pragma unroll
  for (int r = 0; r < 8; ++r) { mi[r] = -1e30f; li[r] = 0.0f; }

  const int vrow = tid >> 3;            // kv row 0..31
  const int vcol = (tid & 7) * 8;       // d 0,8,...,56
  const int kvEnd = blockQ0 + 128;

  for (int j0 = 0; j0 < kvEnd; j0 += 32) {
#if HAVE_TDM
    if (w == 0)
      tdm_load_k_tile(sK, K + base + (size_t)j0 * HD_, (uint32_t)(T_ - j0));
#else
    async_copy16(&sK[vrow * 72 + vcol], K + base + (size_t)(j0 + vrow) * HD_ + vcol);
#endif

    // V tile: global -> regs -> transposed LDS [d][kv]
    v8bf vv = *(const v8bf*)(V + base + (size_t)(j0 + vrow) * HD_ + vcol);
    #pragma unroll
    for (int j = 0; j < 8; ++j) sV[(vcol + j) * 40 + vrow] = vv[j];
    if (j0 + 32 < kvEnd)
      __builtin_prefetch(V + base + (size_t)(j0 + 32 + vrow) * HD_ + vcol, 0, 0);

#if HAVE_TDM
    if (w == 0) __builtin_amdgcn_s_wait_tensorcnt(0);  // TDM writes have landed
#else
    wait_async_0();
#endif
    __syncthreads();         // K tile + V stores visible to all waves

    if (j0 <= waveQ0 + 15) {   // wave-uniform: EXEC stays all-ones inside
      // Batch K fragments, then S = Q K^T (16 q x 32 kv)
      v16bf bk[4];
      #pragma unroll
      for (int nt = 0; nt < 2; ++nt) {
        const __bf16* krow = sK + (nt * 16 + n15) * 72;
        bk[nt * 2 + 0] = frag16(krow +  0 + kb16, krow +  0 + kb16 + 8);
        bk[nt * 2 + 1] = frag16(krow + 32 + kb16, krow + 32 + kb16 + 8);
      }
      v8f s[2] = {};
      s[0] = wmma_bf16(aq0, bk[0], s[0]); s[0] = wmma_bf16(aq1, bk[1], s[0]);
      s[1] = wmma_bf16(aq0, bk[2], s[1]); s[1] = wmma_bf16(aq1, bk[3], s[1]);

      // scale + causal mask + online softmax (16-lane row reductions, wave32)
      float sc[8];
      #pragma unroll
      for (int r = 0; r < 8; ++r) {
        const int qi = waveQ0 + r + half * 8;
        float v0 = (j0 +      n15 <= qi) ? s[0][r] * 0.125f : -1e30f;  // 1/sqrt(64)
        float v1 = (j0 + 16 + n15 <= qi) ? s[1][r] * 0.125f : -1e30f;
        float rm = fmaxf(v0, v1);
        #pragma unroll
        for (int mk = 1; mk < 16; mk <<= 1) rm = fmaxf(rm, __shfl_xor(rm, mk, 32));
        float mn = fmaxf(mi[r], rm);
        sc[r] = __expf(mi[r] - mn);
        mi[r] = mn;
        float p0 = __expf(v0 - mn);
        float p1 = __expf(v1 - mn);
        s[0][r] = p0; s[1][r] = p1;
        float rs = p0 + p1;
        #pragma unroll
        for (int mk = 1; mk < 16; mk <<= 1) rs += __shfl_xor(rs, mk, 32);
        li[r] = li[r] * sc[r] + rs;
      }
      #pragma unroll
      for (int dt = 0; dt < 4; ++dt)
        #pragma unroll
        for (int r = 0; r < 8; ++r) yacc[dt][r] *= sc[r];

      // Stage P (C-layout) -> per-wave LDS [q][kv], then read back as A-fragment
      __bf16* pw = sP + w * (16 * 40);
      #pragma unroll
      for (int r = 0; r < 8; ++r) {
        int row = r + half * 8;
        pw[row * 40 +      n15] = (__bf16)s[0][r];
        pw[row * 40 + 16 + n15] = (__bf16)s[1][r];
      }
      v16bf ap = frag16(pw + n15 * 40 + kb16, pw + n15 * 40 + kb16 + 8);

      // Batch V fragments, then Y += P(16x32) @ V(32x64)
      v16bf bv[4];
      #pragma unroll
      for (int dt = 0; dt < 4; ++dt) {
        const __bf16* vd = sV + (dt * 16 + n15) * 40 + kb16;
        bv[dt] = frag16(vd, vd + 8);
      }
      #pragma unroll
      for (int dt = 0; dt < 4; ++dt)
        yacc[dt] = wmma_bf16(ap, bv[dt], yacc[dt]);
    }
    __syncthreads();
  }

  // Epilogue: normalize and write y in [B,T,C] bf16 layout
  const int b = blockIdx.y / H_;
  const int h = blockIdx.y % H_;
  #pragma unroll
  for (int r = 0; r < 8; ++r) {
    float inv = 1.0f / li[r];
    int t = waveQ0 + r + half * 8;
    __bf16* yrow = Y + ((size_t)(b * T_ + t)) * C_ + h * HD_;
    #pragma unroll
    for (int dt = 0; dt < 4; ++dt)
      yrow[dt * 16 + n15] = (__bf16)(yacc[dt][r] * inv);
  }
}

// ---------------------------------------------------------------------------
// Output projection: out[m, n] = Yb[m, :] . W_proj[n, :] + b_proj[n]  (fp32 out)
// ---------------------------------------------------------------------------
__global__ __launch_bounds__(256) void proj_gemm_kernel(
    const __bf16* __restrict__ Yb, const __bf16* __restrict__ W,
    const float* __restrict__ bias, float* __restrict__ out) {
  __shared__ __bf16 sB[2][64 * 40];

  const int tid  = threadIdx.x;
  const int lane = tid & 31;
  const int w    = tid >> 5;
  const int half = lane >> 4;
  const int n15  = lane & 15;
  const int kb8  = half * 8;
  const int kb16 = half * 16;

  const int m0 = blockIdx.x * 256 + w * 32;
  const int n0 = blockIdx.y * 64;

  const int brow = tid >> 2;
  const int bcol = (tid & 3) * 8;
  const __bf16* wsrc = W + (size_t)(n0 + brow) * C_ + bcol;

  async_copy16(&sB[0][brow * 40 + bcol], wsrc);

  const __bf16* arow0 = Yb + (size_t)(m0 + n15) * C_;
  const __bf16* arow1 = Yb + (size_t)(m0 + 16 + n15) * C_;
  v8f acc[2][4] = {};

  for (int kc = 0; kc < C_; kc += 32) {
    const int cur = (kc >> 5) & 1;
    v16bf a0 = frag16(arow0 + kc + kb16, arow0 + kc + kb16 + 8);
    v16bf a1 = frag16(arow1 + kc + kb16, arow1 + kc + kb16 + 8);
    if (kc + 32 < C_) {
      async_copy16(&sB[cur ^ 1][brow * 40 + bcol], wsrc + kc + 32);
      wait_async_le1();
    } else {
      wait_async_0();
    }
    __syncthreads();
    const __bf16* bb = &sB[cur][0];
    v16bf bf[4];
    #pragma unroll
    for (int nt = 0; nt < 4; ++nt) {
      const __bf16* bp = bb + (nt * 16 + n15) * 40 + kb16;
      bf[nt] = frag16(bp, bp + 8);
    }
    #pragma unroll
    for (int nt = 0; nt < 4; ++nt) {
      acc[0][nt] = wmma_bf16(a0, bf[nt], acc[0][nt]);
      acc[1][nt] = wmma_bf16(a1, bf[nt], acc[1][nt]);
    }
    __syncthreads();
  }

  #pragma unroll
  for (int ms = 0; ms < 2; ++ms) {
    #pragma unroll
    for (int nt = 0; nt < 4; ++nt) {
      int n = n0 + nt * 16 + n15;
      float bv = bias[n];
      #pragma unroll
      for (int r = 0; r < 8; ++r) {
        int m = m0 + ms * 16 + r + half * 8;
        out[(size_t)m * C_ + n] = acc[ms][nt][r] + bv;
      }
    }
  }
}

// ---------------------------------------------------------------------------
extern "C" void kernel_launch(void* const* d_in, const int* in_sizes, int n_in,
                              void* d_out, int out_size, void* d_ws, size_t ws_size,
                              hipStream_t stream) {
  (void)in_sizes; (void)n_in; (void)out_size; (void)ws_size;
  const float* x      = (const float*)d_in[0];
  const float* W_attn = (const float*)d_in[1];
  const float* b_attn = (const float*)d_in[2];
  const float* W_proj = (const float*)d_in[3];
  const float* b_proj = (const float*)d_in[4];
  float* out = (float*)d_out;

  char* ws = (char*)d_ws;
  size_t off = 0;
  auto alloc_bf = [&](size_t nelem) {
    __bf16* p = (__bf16*)(ws + off);
    off += nelem * sizeof(__bf16);
    off = (off + 255) & ~(size_t)255;
    return p;
  };
  const size_t nx  = (size_t)B_ * T_ * C_;
  const size_t nwa = (size_t)3 * C_ * C_;
  const size_t nwp = (size_t)C_ * C_;
  __bf16* xb  = alloc_bf(nx);
  __bf16* wab = alloc_bf(nwa);
  __bf16* wpb = alloc_bf(nwp);
  __bf16* qb  = alloc_bf(nx);
  __bf16* kb  = alloc_bf(nx);
  __bf16* vb  = alloc_bf(nx);
  __bf16* yb  = alloc_bf(nx);

  cvt_bf16_kernel<<<dim3((unsigned)((nx  + 255) / 256)), 256, 0, stream>>>(xb,  x,      (int)nx);
  cvt_bf16_kernel<<<dim3((unsigned)((nwa + 255) / 256)), 256, 0, stream>>>(wab, W_attn, (int)nwa);
  cvt_bf16_kernel<<<dim3((unsigned)((nwp + 255) / 256)), 256, 0, stream>>>(wpb, W_proj, (int)nwp);

  qkv_gemm_kernel<<<dim3(32, 36), 256, 0, stream>>>(xb, wab, b_attn, qb, kb, vb);
  flash_attn_kernel<<<dim3(T_ / 128, B_ * H_), 256, 0, stream>>>(qb, kb, vb, yb);
  proj_gemm_kernel<<<dim3(32, C_ / 64), 256, 0, stream>>>(yb, wpb, b_proj, out);
}
